// ODEBlock_s_43628277792921
// MI455X (gfx1250) — matrix operand order repaired and checked by reading, compiled, not run on hardware
//
#include <hip/hip_runtime.h>
#include <hip/hip_bf16.h>
#include <math.h>

// ---------------------------------------------------------------------------
// Types for CDNA5 WMMA
// ---------------------------------------------------------------------------
typedef __attribute__((ext_vector_type(16))) __bf16 v16bf;
typedef __attribute__((ext_vector_type(8)))  float  v8f;

#define TILE_M 256
#define TILE_N 64
#define TILE_K 32
#define BPAD   16                 // B row padding: 48 elems = 96B (32B aligned)
#define THREADS 256

__device__ __forceinline__ void bsplit(float x, __bf16& h, __bf16& l) {
    h = (__bf16)x;
    l = (__bf16)(x - (float)h);
}

union UA { uint4 u; __bf16 b[8]; };
union UB { uint2 u; __bf16 b[4]; };

// ---------------------------------------------------------------------------
// GEMM: C[M,N] = (Ahi+Alo)[M,K] @ (Bhi+Blo)[K,N] + bias, fp32 accumulation via
// 3-pass bf16 WMMA (hi*hi + hi*lo + lo*hi).
// - A staged in LDS with k-segments permuted {0,2,1,3} so each lane's fragment
//   is one contiguous 32B v16bf load.
// - B staged transposed [n][k] for the same reason.
// - Software-pipelined K loop over a 2-deep LDS buffer: chunk i+1 is
//   global_prefetch'ed (global_prefetch_b8) before the WMMA burst on chunk i,
//   then staged global->LDS in the handoff phase (loads hit near caches).
//   No staging registers are held across the compute phase -> no spills.
// Each wave computes a 32x64 strip (8 accumulators).
// EPI==1: out = tanh(acc+bias), re-split to bf16 hi/lo (layer-1 epilogue)
// EPI==0: out = acc+bias as fp32                         (layer-2 epilogue)
// ---------------------------------------------------------------------------
template <int EPI>
__global__ __launch_bounds__(THREADS, 1)
void gemm_bf16split_kernel(const __bf16* __restrict__ Ahi,
                           const __bf16* __restrict__ Alo,
                           const __bf16* __restrict__ Bhi,
                           const __bf16* __restrict__ Blo,
                           const float*  __restrict__ bias,
                           float*  __restrict__ outF,
                           __bf16* __restrict__ outHi,
                           __bf16* __restrict__ outLo,
                           int M, int N, int K) {
    __shared__ __align__(32) __bf16 sAhi[2][TILE_M][TILE_K];
    __shared__ __align__(32) __bf16 sAlo[2][TILE_M][TILE_K];
    __shared__ __align__(32) __bf16 sBThi[2][TILE_N][TILE_K + BPAD];
    __shared__ __align__(32) __bf16 sBTlo[2][TILE_N][TILE_K + BPAD];

    const int tid  = threadIdx.x;
    const int lane = tid & 31;
    const int wv   = tid >> 5;          // 8 waves: wave w owns rows [32w,32w+32)
    const int lh   = lane & 15;
    const int hiHalf = (lane >= 16) ? 1 : 0;

    const int m0 = blockIdx.y * TILE_M;
    const int n0 = blockIdx.x * TILE_N;

    // B staging role of this thread (threads 0..127 stage B)
    const int bplane = tid >> 6;              // 0 = hi, 1 = lo
    const int bt     = tid & 63;
    const int bkk    = (bt >> 3) * 4;         // k within chunk
    const int bnn    = (bt & 7) * 8;          // n within tile
    const bool doB   = (tid < 128);

    // prefetch next chunk (no VGPR data, no counters): one cacheline per row
    auto prefetch_chunk = [&](int k0) {
        __builtin_prefetch(Ahi + (size_t)(m0 + tid) * K + k0, 0, 3);
        __builtin_prefetch(Alo + (size_t)(m0 + tid) * K + k0, 0, 3);
        if (tid < 32)
            __builtin_prefetch(Bhi + (size_t)(k0 + tid) * N + n0, 0, 3);
        else if (tid < 64)
            __builtin_prefetch(Blo + (size_t)(k0 + (tid - 32)) * N + n0, 0, 3);
    };

    // stage chunk at k0 into LDS buffer `buf` (global loads -> LDS stores)
    auto stage = [&](int buf, int k0) {
#pragma unroll
        for (int it = 0; it < 4; ++it) {
            const int i    = tid + it * THREADS;     // 0..1023
            const int row  = i >> 2, seg = i & 3;
            const int phys = ((seg & 1) << 1) | (seg >> 1);  // 0->0,1->2,2->1,3->3
            uint4 vh = *((const uint4*)(Ahi + (size_t)(m0 + row) * K + k0) + seg);
            uint4 vl = *((const uint4*)(Alo + (size_t)(m0 + row) * K + k0) + seg);
            *((uint4*)&sAhi[buf][row][phys * 8]) = vh;
            *((uint4*)&sAlo[buf][row][phys * 8]) = vl;
        }
        if (doB) {
            const __bf16* __restrict__ src = bplane ? Blo : Bhi;
            UA r0, r1, r2, r3;
            r0.u = *((const uint4*)(src + (size_t)(k0 + bkk + 0) * N + n0 + bnn));
            r1.u = *((const uint4*)(src + (size_t)(k0 + bkk + 1) * N + n0 + bnn));
            r2.u = *((const uint4*)(src + (size_t)(k0 + bkk + 2) * N + n0 + bnn));
            r3.u = *((const uint4*)(src + (size_t)(k0 + bkk + 3) * N + n0 + bnn));
#pragma unroll
            for (int q = 0; q < 8; ++q) {
                UB w;
                w.b[0] = r0.b[q]; w.b[1] = r1.b[q];
                w.b[2] = r2.b[q]; w.b[3] = r3.b[q];
                __bf16* dst = bplane ? &sBTlo[buf][bnn + q][bkk]
                                     : &sBThi[buf][bnn + q][bkk];
                *((uint2*)dst) = w.u;
            }
        }
    };

    v8f acc[2][4];
#pragma unroll
    for (int ms = 0; ms < 2; ++ms)
#pragma unroll
        for (int j = 0; j < 4; ++j) acc[ms][j] = (v8f)0.0f;

    const int nck = K / TILE_K;

    // ---- pipeline prologue: stage chunk 0 ----
    stage(0, 0);
    __syncthreads();

    int buf = 0;
    for (int ck = 0; ck < nck; ++ck) {
        // pull chunk i+1 toward the caches while the matrix pipes run
        if (ck + 1 < nck) prefetch_chunk((ck + 1) * TILE_K);

        // ---- compute on LDS[buf] ----
        const __bf16 (*__restrict__ pAh)[TILE_K]        = sAhi[buf];
        const __bf16 (*__restrict__ pAl)[TILE_K]        = sAlo[buf];
        const __bf16 (*__restrict__ pBh)[TILE_K + BPAD] = sBThi[buf];
        const __bf16 (*__restrict__ pBl)[TILE_K + BPAD] = sBTlo[buf];

        v16bf ah[2], al[2];
#pragma unroll
        for (int ms = 0; ms < 2; ++ms) {
            const int mrow = wv * 32 + ms * 16 + lh;
            ah[ms] = *((const v16bf*)&pAh[mrow][hiHalf * 16]);
            al[ms] = *((const v16bf*)&pAl[mrow][hiHalf * 16]);
        }
#pragma unroll
        for (int j = 0; j < 4; ++j) {
            const int ncol = j * 16 + lh;
            typedef __attribute__((ext_vector_type(8))) __bf16 v8bf_t;
            v8bf_t bh0 = *((const v8bf_t*)&pBh[ncol][hiHalf * 8]);
            v8bf_t bh1 = *((const v8bf_t*)&pBh[ncol][16 + hiHalf * 8]);
            v8bf_t bl0 = *((const v8bf_t*)&pBl[ncol][hiHalf * 8]);
            v8bf_t bl1 = *((const v8bf_t*)&pBl[ncol][16 + hiHalf * 8]);
            v16bf bh = __builtin_shufflevector(bh0, bh1, 0,1,2,3,4,5,6,7,8,9,10,11,12,13,14,15);
            v16bf bl = __builtin_shufflevector(bl0, bl1, 0,1,2,3,4,5,6,7,8,9,10,11,12,13,14,15);
#pragma unroll
            for (int ms = 0; ms < 2; ++ms) {
                acc[ms][j] = __builtin_amdgcn_wmma_f32_16x16x32_bf16(
                    false, ah[ms], false, bh, (short)0, acc[ms][j], false, false);
                acc[ms][j] = __builtin_amdgcn_wmma_f32_16x16x32_bf16(
                    false, ah[ms], false, bl, (short)0, acc[ms][j], false, false);
                acc[ms][j] = __builtin_amdgcn_wmma_f32_16x16x32_bf16(
                    false, al[ms], false, bh, (short)0, acc[ms][j], false, false);
            }
        }

        // ---- hand off next chunk to the other LDS buffer ----
        if (ck + 1 < nck) {
            stage(buf ^ 1, (ck + 1) * TILE_K);
            __syncthreads();
            buf ^= 1;
        }
    }

    // ---- epilogue: C layout = VGPR r -> M = r + 8*hiHalf, N = lane&15 ----
#pragma unroll
    for (int j = 0; j < 4; ++j) {
        const int col = n0 + j * 16 + lh;
        const float bv = bias[col];
#pragma unroll
        for (int ms = 0; ms < 2; ++ms) {
#pragma unroll
            for (int r = 0; r < 8; ++r) {
                const int row = m0 + wv * 32 + ms * 16 + r + (hiHalf ? 8 : 0);
                const float v = acc[ms][j][r] + bv;
                if (EPI == 1) {
                    const float t = tanhf(v);
                    __bf16 h, l;
                    bsplit(t, h, l);
                    outHi[(size_t)row * N + col] = h;
                    outLo[(size_t)row * N + col] = l;
                } else {
                    outF[(size_t)row * N + col] = v;
                }
            }
        }
    }
}

// ---------------------------------------------------------------------------
// Elementwise helpers
// ---------------------------------------------------------------------------
__global__ void split_kernel(const float* __restrict__ x,
                             __bf16* __restrict__ hi, __bf16* __restrict__ lo,
                             int n) {
    for (int i = blockIdx.x * blockDim.x + threadIdx.x; i < n;
         i += gridDim.x * blockDim.x) {
        __bf16 h, l;
        bsplit(x[i], h, l);
        hi[i] = h; lo[i] = l;
    }
}

// out = y + a*k ; split(out) -> hi/lo    (in-place out==y allowed)
__global__ void axpy_split_kernel(const float* __restrict__ y,
                                  const float* __restrict__ k, float a,
                                  float* __restrict__ out,
                                  __bf16* __restrict__ hi, __bf16* __restrict__ lo,
                                  int n) {
    for (int i = blockIdx.x * blockDim.x + threadIdx.x; i < n;
         i += gridDim.x * blockDim.x) {
        float v = y[i] + a * k[i];
        out[i] = v;
        __bf16 h, l;
        bsplit(v, h, l);
        hi[i] = h; lo[i] = l;
    }
}

// acc = beta*acc + a*f
__global__ void acc_axpy_kernel(float* __restrict__ acc,
                                const float* __restrict__ f,
                                float a, float beta, int n) {
    for (int i = blockIdx.x * blockDim.x + threadIdx.x; i < n;
         i += gridDim.x * blockDim.x) {
        acc[i] = beta * acc[i] + a * f[i];
    }
}

// out = y + c0*a0 + c1*a1 + c2*a2 + c3*a3 ; split(out)
__global__ void comb4_split_kernel(const float* __restrict__ y,
                                   const float* __restrict__ a0,
                                   const float* __restrict__ a1,
                                   const float* __restrict__ a2,
                                   const float* __restrict__ a3,
                                   float c0, float c1, float c2, float c3,
                                   float* __restrict__ out,
                                   __bf16* __restrict__ hi, __bf16* __restrict__ lo,
                                   int n) {
    for (int i = blockIdx.x * blockDim.x + threadIdx.x; i < n;
         i += gridDim.x * blockDim.x) {
        float v = y[i] + c0 * a0[i] + c1 * a1[i] + c2 * a2[i] + c3 * a3[i];
        out[i] = v;
        __bf16 h, l;
        bsplit(v, h, l);
        hi[i] = h; lo[i] = l;
    }
}

// ---------------------------------------------------------------------------
// Host orchestration
// ---------------------------------------------------------------------------
extern "C" void kernel_launch(void* const* d_in, const int* in_sizes, int n_in,
                              void* d_out, int out_size, void* d_ws, size_t ws_size,
                              hipStream_t stream) {
    const int Bn = 4096, Dn = 512, Hn = 2048;
    const int N_STEPS = 20, N_CORR = 3;
    const float dt = 1.0f / N_STEPS;
    const float c  = dt / 24.0f;

    const float* x  = (const float*)d_in[0];
    const float* W1 = (const float*)d_in[1];
    const float* b1 = (const float*)d_in[2];
    const float* W2 = (const float*)d_in[3];
    const float* b2 = (const float*)d_in[4];

    // ---- bump-allocate workspace ----
    char* p = (char*)d_ws;
    auto alloc = [&](size_t bytes) -> char* {
        char* r = p;
        p += (bytes + 255) & ~(size_t)255;
        return r;
    };
    const size_t nBD = (size_t)Bn * Dn;   // 2M elements
    const size_t nBH = (size_t)Bn * Hn;   // 8M elements
    const size_t nDH = (size_t)Dn * Hn;   // 1M elements

    __bf16* W1hi = (__bf16*)alloc(nDH * 2);
    __bf16* W1lo = (__bf16*)alloc(nDH * 2);
    __bf16* W2hi = (__bf16*)alloc(nDH * 2);
    __bf16* W2lo = (__bf16*)alloc(nDH * 2);
    __bf16* inhi = (__bf16*)alloc(nBD * 2);
    __bf16* inlo = (__bf16*)alloc(nBD * 2);
    __bf16* Zhi  = (__bf16*)alloc(nBH * 2);
    __bf16* Zlo  = (__bf16*)alloc(nBH * 2);
    float*  y    = (float*)alloc(nBD * 4);
    float*  S    = (float*)alloc(nBD * 4);
    float*  F    = (float*)alloc(nBD * 4);
    float*  acc  = (float*)alloc(nBD * 4);
    float*  hbuf[4];
    for (int i = 0; i < 4; ++i) hbuf[i] = (float*)alloc(nBD * 4);
    (void)ws_size; (void)in_sizes; (void)n_in;

    const int EW_BLOCKS = 1024, EW_THREADS = 256;
    const int nBDi = (int)nBD;

    // ---- split weights into bf16 hi/lo (once per launch, deterministic) ----
    split_kernel<<<EW_BLOCKS, EW_THREADS, 0, stream>>>(W1, W1hi, W1lo, (int)nDH);
    split_kernel<<<EW_BLOCKS, EW_THREADS, 0, stream>>>(W2, W2hi, W2lo, (int)nDH);

    // ---- y = x ; split(x) as first f input ----
    hipMemcpyAsync(y, x, nBD * 4, hipMemcpyDeviceToDevice, stream);
    split_kernel<<<EW_BLOCKS, EW_THREADS, 0, stream>>>(x, inhi, inlo, nBDi);

    // f(.) : gemm1 (tanh-split epilogue) then gemm2 (fp32 + bias) -> dst
    auto f_eval = [&](float* dst) {
        dim3 g1(Hn / TILE_N, Bn / TILE_M);
        gemm_bf16split_kernel<1><<<g1, THREADS, 0, stream>>>(
            inhi, inlo, W1hi, W1lo, b1, nullptr, Zhi, Zlo, Bn, Hn, Dn);
        dim3 g2(Dn / TILE_N, Bn / TILE_M);
        gemm_bf16split_kernel<0><<<g2, THREADS, 0, stream>>>(
            Zhi, Zlo, W2hi, W2lo, b2, dst, nullptr, nullptr, Bn, Dn, Hn);
    };

    // ---- hist[0] = f(x) ----
    f_eval(hbuf[0]);

    // ---- 3 RK4 bootstrap steps; k1 reuses newest hist entry = f(y) ----
    for (int s = 0; s < 3; ++s) {
        float* k1 = hbuf[s];
        acc_axpy_kernel<<<EW_BLOCKS, EW_THREADS, 0, stream>>>(acc, k1, 1.0f, 0.0f, nBDi);
        axpy_split_kernel<<<EW_BLOCKS, EW_THREADS, 0, stream>>>(y, k1, 0.5f * dt, S, inhi, inlo, nBDi);
        f_eval(F);                                                        // k2
        acc_axpy_kernel<<<EW_BLOCKS, EW_THREADS, 0, stream>>>(acc, F, 2.0f, 1.0f, nBDi);
        axpy_split_kernel<<<EW_BLOCKS, EW_THREADS, 0, stream>>>(y, F, 0.5f * dt, S, inhi, inlo, nBDi);
        f_eval(F);                                                        // k3
        acc_axpy_kernel<<<EW_BLOCKS, EW_THREADS, 0, stream>>>(acc, F, 2.0f, 1.0f, nBDi);
        axpy_split_kernel<<<EW_BLOCKS, EW_THREADS, 0, stream>>>(y, F, dt, S, inhi, inlo, nBDi);
        f_eval(F);                                                        // k4
        acc_axpy_kernel<<<EW_BLOCKS, EW_THREADS, 0, stream>>>(acc, F, 1.0f, 1.0f, nBDi);
        // y += dt/6 * acc, split(y) for the hist evaluation / next k1
        axpy_split_kernel<<<EW_BLOCKS, EW_THREADS, 0, stream>>>(y, acc, dt / 6.0f, y, inhi, inlo, nBDi);
        f_eval(hbuf[s + 1]);                                              // hist push-front
    }

    // hist newest -> oldest
    float* h0 = hbuf[3];
    float* h1 = hbuf[2];
    float* h2 = hbuf[1];
    float* h3 = hbuf[0];

    // ---- 17 ABM4 predictor/corrector steps ----
    for (int s = 0; s < N_STEPS - 3; ++s) {
        // predictor: S = y + c*(55 h0 - 59 h1 + 37 h2 - 9 h3)
        comb4_split_kernel<<<EW_BLOCKS, EW_THREADS, 0, stream>>>(
            y, h0, h1, h2, h3, 55.0f * c, -59.0f * c, 37.0f * c, -9.0f * c,
            S, inhi, inlo, nBDi);
        for (int it = 0; it < N_CORR; ++it) {
            f_eval(F);  // f(pred)
            // corrector: S = y + c*(9 F + 19 h0 - 5 h1 + h2)
            comb4_split_kernel<<<EW_BLOCKS, EW_THREADS, 0, stream>>>(
                y, F, h0, h1, h2, 9.0f * c, 19.0f * c, -5.0f * c, c,
                S, inhi, inlo, nBDi);
        }
        f_eval(h3);               // f_new = f(pred) into oldest hist buffer
        float* hn = h3;           // rotate hist: [f_new, h0, h1, h2]
        h3 = h2; h2 = h1; h1 = h0; h0 = hn;
        float* t = y; y = S; S = t;   // y = pred
    }

    hipMemcpyAsync(d_out, y, (size_t)out_size * 4, hipMemcpyDeviceToDevice, stream);
}
